// TimeAttentionGRU_19000935318215
// MI455X (gfx1250) — matrix operand order repaired and checked by reading, not compiled, not run on hardware
//
#include <hip/hip_runtime.h>

typedef __attribute__((ext_vector_type(16))) _Float16 v16h;
typedef __attribute__((ext_vector_type(8)))  _Float16 v8h;
typedef __attribute__((ext_vector_type(8)))  float    v8f;
typedef __attribute__((ext_vector_type(4)))  float    v4f;

union AF { v16h v; v8h h2[2]; _Float16 e[16]; };

#define S_LEN 4096
#define B_SZ  32
#define D_IN  64
#define H_DIM 128
#define CG    48   // C / GROUPS
#define ROWP  200  // padded row stride (halves) of activation buffers
#define ABUF  (16 * ROWP + 16)   // +16 halves guard so K-padding reads stay in-array
#define GATE_ELEMS (3*8*2*32*16) // 3 gates, 8 n-tiles, 2 K-chunks, 32 lanes, 16 f16
#define W1_ELEMS   (4*8*32*16)   // 4 K-chunks, 8 n-tiles

// Fast activations: single v_exp + v_rcp, correct saturation at +-inf.
__device__ __forceinline__ float fast_sigmoid(float x) {
  return __builtin_amdgcn_rcpf(1.f + __expf(-x));
}
__device__ __forceinline__ float fast_tanh(float x) {
  return 1.f - 2.f * __builtin_amdgcn_rcpf(1.f + __expf(2.f * x));
}

// ---------------------------------------------------------------------------
// Kernel 0: pack center-tap gate weights + W1 into WMMA B-fragment lane order.
// Lane L: Nl = L%16 (column), halfK = L/16; element e: K = 32*chunk+8*halfK+(e&7)+16*(e>>3).
// Gate fragments are ZERO for K >= 48: this both implements the K=48->64 pad
// and annihilates whatever the A fragment reads in its padding region.
// ---------------------------------------------------------------------------
__global__ __launch_bounds__(256) void prep_weights(
    const float* __restrict__ Wz, const float* __restrict__ Wr,
    const float* __restrict__ Wh, const float* __restrict__ W1,
    _Float16* __restrict__ gateF, _Float16* __restrict__ w1F)
{
  int i = blockIdx.x * 256 + threadIdx.x;
  if (i < GATE_ELEMS) {
    int e = i & 15, lane = (i >> 4) & 31, chunk = (i >> 9) & 1;
    int n = (i >> 10) & 7, gate = i >> 13;
    int Nl = lane & 15, halfK = lane >> 4;
    int K = 32 * chunk + 8 * halfK + (e & 7) + 16 * (e >> 3);      // padded K in [0,64)
    int h = 16 * n + Nl;                      // output channel
    float val = 0.f;
    if (K < CG) {
      const float* W = (gate == 0) ? Wz : ((gate == 1) ? Wr : Wh);
      val = W[h * (CG * 5) + K * 5 + 2];      // center tap of (H, C/G, 5)
    }
    gateF[i] = (_Float16)val;
  } else {
    int j = i - GATE_ELEMS;
    if (j < W1_ELEMS) {
      int e = j & 15, lane = (j >> 4) & 31, n = (j >> 9) & 7, chunk = j >> 12;
      int Nl = lane & 15, halfK = lane >> 4;
      int K = 32 * chunk + 8 * halfK + (e & 7) + 16 * (e >> 3);    // in [0,128)
      int col = 16 * n + Nl;
      w1F[j] = (_Float16)W1[K * H_DIM + col];
    }
  }
}

// ---------------------------------------------------------------------------
// Kernel 1: sequential GRU scan. 2 blocks (16 batch rows each), 8 waves,
// wave n owns h columns [16n, 16n+16). Weights live in registers across all
// 4096 steps; h state f32 in LDS; 2 barriers per step. A fragments are four
// straight ds_load_b128 per phase (no per-step zero fills / register copies);
// the K>=48 garbage is multiplied by zeroed weight rows.
// ---------------------------------------------------------------------------
__global__ __launch_bounds__(256) void gru_scan(
    const float* __restrict__ x,
    const float* __restrict__ bz, const float* __restrict__ br,
    const float* __restrict__ bh,
    const _Float16* __restrict__ gateF,
    _Float16* __restrict__ g)
{
  __shared__ __align__(16) _Float16 comb [ABUF];  // [row][192ch] pad->200 +guard
  __shared__ __align__(16) _Float16 comb2[ABUF];
  __shared__ float hbuf[16 * 132];                // f32 h state

  const int tid  = threadIdx.x;
  const int lane = tid & 31;
  const int n    = tid >> 5;        // wave = n-tile 0..7
  const int grp  = n >> 1;          // group 0..3
  const int Nl   = lane & 15;
  const int half = lane >> 4;
  const int b0   = blockIdx.x * 16;

  for (int i = tid; i < ABUF; i += 256) { comb[i] = (_Float16)0.f; comb2[i] = (_Float16)0.f; }
  for (int i = tid; i < 16 * 132; i += 256) hbuf[i] = 0.f;

  const int col  = 16 * n + Nl;     // this lane's output h channel
  const float bzc = bz[col], brc = br[col], bhc = bh[col];

  // Weight B-fragments, resident in registers for the whole scan.
  const v16h* wzp = (const v16h*)(gateF + (0 * 8 + n) * 1024);
  const v16h* wrp = (const v16h*)(gateF + (1 * 8 + n) * 1024);
  const v16h* whp = (const v16h*)(gateF + (2 * 8 + n) * 1024);
  v16h wz0 = wzp[lane], wz1 = wzp[32 + lane];
  v16h wr0 = wrp[lane], wr1 = wrp[32 + lane];
  v16h wh0 = whp[lane], wh1 = whp[32 + lane];

  // x staging: thread owns 4 consecutive floats of the 16x64 step tile.
  const int xrow = tid >> 4;
  const int xc   = (tid & 15) << 2;
  const float* xbase = x + ((size_t)(b0 + xrow) * S_LEN) * D_IN + xc;
  v4f xcur  = *(const v4f*)(xbase);            // x_0
  v4f xnext = *(const v4f*)(xbase + D_IN);     // x_1

  // g streaming: thread copies 8 contiguous f16 of one h row.
  const int grow = tid >> 4;
  const int gc8  = (tid & 15) * 8;

  { _Float16* cx = comb + xrow * ROWP + xc;    // comb.x = x_0 (h already 0)
    #pragma unroll
    for (int k = 0; k < 4; ++k) cx[k] = (_Float16)xcur[k]; }
  __syncthreads();

  const int fragoff = Nl * ROWP + grp * CG + 8 * half;  // A-frag base (halves)

  for (int t = 0; t < S_LEN; ++t) {
    // ---------------- phase 1 ----------------
    if (t > 0) {   // stream h_t = g[.][t-1], coalesced b128 per thread
      v8h hv = *(const v8h*)(comb + grow * ROWP + 64 + gc8);
      *(v8h*)(g + ((size_t)(b0 + grow) * S_LEN + (t - 1)) * H_DIM + gc8) = hv;
    }
    { _Float16* cx2 = comb2 + xrow * ROWP + xc; // comb2.x = x_t
      #pragma unroll
      for (int k = 0; k < 4; ++k) cx2[k] = (_Float16)xcur[k]; }

    const _Float16* rp = comb + fragoff;
    AF a0, a1;
    a0.h2[0] = *(const v8h*)(rp);
    a0.h2[1] = *(const v8h*)(rp + 16);
    a1.h2[0] = *(const v8h*)(rp + 32);
    a1.h2[1] = *(const v8h*)(rp + 48);   // K>=48: finite junk x 0-weights

    v8f accz = {};
    accz = __builtin_amdgcn_wmma_f32_16x16x32_f16(false, a0.v, false, wz0, (short)0, accz, false, false);
    accz = __builtin_amdgcn_wmma_f32_16x16x32_f16(false, a1.v, false, wz1, (short)0, accz, false, false);
    v8f accr = {};
    accr = __builtin_amdgcn_wmma_f32_16x16x32_f16(false, a0.v, false, wr0, (short)0, accr, false, false);
    accr = __builtin_amdgcn_wmma_f32_16x16x32_f16(false, a1.v, false, wr1, (short)0, accr, false, false);

    float zf[8], hf[8];
    #pragma unroll
    for (int j = 0; j < 8; ++j) {
      int Mr = j + 8 * half;                              // batch row (local)
      float zv = fast_sigmoid(accz[j] + bzc);
      float rv = fast_sigmoid(accr[j] + brc);
      float hv = hbuf[Mr * 132 + col];
      zf[j] = zv; hf[j] = hv;
      comb2[Mr * ROWP + 64 + col] = (_Float16)(rv * hv);  // r * h
    }
    __syncthreads();   // barrier A

    // ---------------- phase 2 ----------------
    const _Float16* rp2 = comb2 + fragoff;
    AF c0, c1;
    c0.h2[0] = *(const v8h*)(rp2);
    c0.h2[1] = *(const v8h*)(rp2 + 16);
    c1.h2[0] = *(const v8h*)(rp2 + 32);
    c1.h2[1] = *(const v8h*)(rp2 + 48);

    v8f acch = {};
    acch = __builtin_amdgcn_wmma_f32_16x16x32_f16(false, c0.v, false, wh0, (short)0, acch, false, false);
    acch = __builtin_amdgcn_wmma_f32_16x16x32_f16(false, c1.v, false, wh1, (short)0, acch, false, false);

    #pragma unroll
    for (int j = 0; j < 8; ++j) {
      int Mr = j + 8 * half;
      float hh = fast_tanh(acch[j] + bhc);
      float hn = hf[j] + zf[j] * (hh - hf[j]);            // (1-z)h + z*h_hat
      hbuf[Mr * 132 + col] = hn;
      comb[Mr * ROWP + 64 + col] = (_Float16)hn;          // h_{t+1}
    }
    if (t + 1 < S_LEN) {                                  // comb.x = x_{t+1}
      _Float16* cx = comb + xrow * ROWP + xc;
      #pragma unroll
      for (int k = 0; k < 4; ++k) cx[k] = (_Float16)xnext[k];
    }
    xcur = xnext;
    if (t + 2 < S_LEN) xnext = *(const v4f*)(xbase + (size_t)(t + 2) * D_IN);
    __syncthreads();   // barrier B
  }
  // final state h_S = g[.][S-1]
  { v8h hv = *(const v8h*)(comb + grow * ROWP + 64 + gc8);
    *(v8h*)(g + ((size_t)(b0 + grow) * S_LEN + (S_LEN - 1)) * H_DIM + gc8) = hv; }
}

// ---------------------------------------------------------------------------
// Kernel 2: scores[m] = tanh(g[m,:] @ W1 + b1) @ W2 + b2, m = b*S + s.
// One block per 16-row M tile; wave n computes a 16x16 tile of g@W1 with
// 4 WMMAs (K=128), then deterministic LDS reduction over the 128 h-columns.
// ---------------------------------------------------------------------------
__global__ __launch_bounds__(256) void scores_kernel(
    const _Float16* __restrict__ g, const _Float16* __restrict__ w1F,
    const float* __restrict__ b1, const float* __restrict__ W2,
    const float* __restrict__ b2, float* __restrict__ scores)
{
  __shared__ float part[256 * 8];
  const int tid = threadIdx.x, lane = tid & 31, n = tid >> 5;
  const int Nl = lane & 15, half = lane >> 4;
  const int mt = blockIdx.x;

  const int row = mt * 16 + Nl;                 // A-fragment M = lane%16
  const _Float16* gr = g + (size_t)row * H_DIM + 8 * half;

  v8f acc = {};
  #pragma unroll
  for (int c = 0; c < 4; ++c) {
    AF a;
    a.h2[0] = *(const v8h*)(gr + 32 * c);
    a.h2[1] = *(const v8h*)(gr + 32 * c + 16);
    v16h w = ((const v16h*)w1F)[(c * 8 + n) * 32 + lane];
    acc = __builtin_amdgcn_wmma_f32_16x16x32_f16(false, a.v, false, w, (short)0, acc, false, false);
  }
  const int col = 16 * n + Nl;
  const float b1c = b1[col], w2c = W2[col];
  #pragma unroll
  for (int j = 0; j < 8; ++j)
    part[tid * 8 + j] = fast_tanh(acc[j] + b1c) * w2c;
  __syncthreads();

  if (tid < 16) {                               // deterministic reduction
    int hh = tid >> 3, j = tid & 7;
    float s = 0.f;
    for (int w = 0; w < 8; ++w)
      for (int l = 0; l < 16; ++l)
        s += part[(w * 32 + hh * 16 + l) * 8 + j];
    scores[mt * 16 + tid] = s + b2[0];
  }
}

// ---------------------------------------------------------------------------
// Kernel 3: per-batch softmax over S and context = sum_s attn * g.
// exp weights cached in LDS so the context pass does no transcendentals.
// ---------------------------------------------------------------------------
__global__ __launch_bounds__(256) void softmax_ctx(
    const float* __restrict__ scores, const _Float16* __restrict__ g,
    float* __restrict__ out)
{
  __shared__ float wbuf[S_LEN];     // 16 KB
  __shared__ float red[256];
  const int b = blockIdx.x, tid = threadIdx.x;
  const float* sb = scores + (size_t)b * S_LEN;

  float mx = -3.4e38f;
  for (int i = tid; i < S_LEN; i += 256) mx = fmaxf(mx, sb[i]);
  red[tid] = mx; __syncthreads();
  for (int s = 128; s > 0; s >>= 1) { if (tid < s) red[tid] = fmaxf(red[tid], red[tid + s]); __syncthreads(); }
  mx = red[0]; __syncthreads();

  float sum = 0.f;
  for (int i = tid; i < S_LEN; i += 256) {
    float w = __expf(sb[i] - mx);
    wbuf[i] = w;
    sum += w;
  }
  red[tid] = sum; __syncthreads();
  for (int s = 128; s > 0; s >>= 1) { if (tid < s) red[tid] += red[tid + s]; __syncthreads(); }
  const float Z = red[0]; __syncthreads();

  const int h = tid & 127, sg = tid >> 7;
  float acc = 0.f;
  for (int s = sg; s < S_LEN; s += 2)
    acc += wbuf[s] * (float)g[((size_t)b * S_LEN + s) * H_DIM + h];
  red[tid] = acc; __syncthreads();
  if (tid < 128) out[b * H_DIM + tid] = (red[tid] + red[tid + 128]) / Z;
}

// ---------------------------------------------------------------------------
extern "C" void kernel_launch(void* const* d_in, const int* in_sizes, int n_in,
                              void* d_out, int out_size, void* d_ws, size_t ws_size,
                              hipStream_t stream) {
  (void)in_sizes; (void)n_in; (void)out_size; (void)ws_size;
  const float* x  = (const float*)d_in[0];
  const float* Wz = (const float*)d_in[1];
  const float* bz = (const float*)d_in[2];
  const float* Wr = (const float*)d_in[3];
  const float* br = (const float*)d_in[4];
  const float* Wh = (const float*)d_in[5];
  const float* bh = (const float*)d_in[6];
  const float* W1 = (const float*)d_in[7];
  const float* b1 = (const float*)d_in[8];
  const float* W2 = (const float*)d_in[9];
  const float* b2 = (const float*)d_in[10];

  char* ws = (char*)d_ws;
  float*     scores = (float*)ws;                                 // 512 KB
  _Float16*  gateF  = (_Float16*)(ws + (512 << 10));              // 48 KB
  _Float16*  w1F    = (_Float16*)(ws + (512 << 10) + (48 << 10)); // 32 KB
  _Float16*  g      = (_Float16*)(ws + (1 << 20));                // 32 MB

  prep_weights<<<(GATE_ELEMS + W1_ELEMS + 255) / 256, 256, 0, stream>>>(
      Wz, Wr, Wh, W1, gateF, w1F);
  gru_scan<<<2, 256, 0, stream>>>(x, bz, br, bh, gateF, g);
  scores_kernel<<<(B_SZ * S_LEN) / 16, 256, 0, stream>>>(g, w1F, b1, W2, b2, scores);
  softmax_ctx<<<B_SZ, 256, 0, stream>>>(scores, g, (float*)d_out);
}
